// EncoderLayer_1391569404196
// MI455X (gfx1250) — matrix operand order repaired and checked by reading, not compiled, not run on hardware
//
#include <hip/hip_runtime.h>

// ---------------------------------------------------------------------------
// Types for CDNA5 WMMA
// ---------------------------------------------------------------------------
typedef __attribute__((ext_vector_type(16))) __bf16 v16bf;
typedef __attribute__((ext_vector_type(8)))  __bf16 v8bf;
typedef __attribute__((ext_vector_type(8)))  float  v8f;

#define D_MODEL 1024
#define SEQ     2048
#define BATCH   2
#define HEADS   16
#define DEPTH   64
#define DFF     4096
#define MROWS   (BATCH * SEQ)   // 4096

__device__ __forceinline__ v8f vzero8() {
    v8f z;
#pragma unroll
    for (int i = 0; i < 8; ++i) z[i] = 0.0f;
    return z;
}

// Load a 16-element bf16 fragment from two 16-byte chunks (A-frag: p1 = p0+16
// elements for the K+16 half; B-frag: likewise p1 = p0+16).
__device__ __forceinline__ v16bf load2x8(const __bf16* p0, const __bf16* p1) {
    union { v16bf v; v8bf h[2]; } u;
    u.h[0] = *(const v8bf*)p0;
    u.h[1] = *(const v8bf*)p1;
    return u.v;
}

// ---------------------------------------------------------------------------
// GEMM: C[M,N] = act(A[M,K] @ W[K,N] + bias), fp32 in/out, bf16 WMMA inside.
// Block = 256 threads = 8 waves, block tile 128x128, K-step 64, double-
// buffered LDS: 16 WMMA per single barrier, next tile's global loads issued
// before current tile's WMMAs (latency hiding).
// ---------------------------------------------------------------------------
__global__ __launch_bounds__(256)
void gemm_bf16_kernel(const float* __restrict__ A, const float* __restrict__ W,
                      const float* __restrict__ bias, float* __restrict__ C,
                      int M, int N, int K, int relu)
{
    __shared__ __align__(16) __bf16 As[2][128 * 72];  // [m][k], stride 72
    __shared__ __align__(16) __bf16 Bs[2][128 * 72];  // [n][k] (transposed)

    const int tid  = threadIdx.x;
    const int lane = tid & 31;
    const int wave = tid >> 5;
    const int wm   = wave & 3;   // 0..3 -> 32-row slabs
    const int wn   = wave >> 2;  // 0..1 -> 64-col slabs
    const int bm   = blockIdx.x * 128;
    const int bn   = blockIdx.y * 128;
    const int half = lane >> 4;
    const int l15  = lane & 15;
    const int nk   = K >> 6;

    // staging coordinates (fixed per thread)
    const int arow = tid >> 4;           // A: 16 float4 per 64-wide row -> rows 0..15 (+16*i)
    const int akc  = (tid & 15) * 4;
    const int bkr  = tid >> 5;           // B: 32 float4 per 128-wide row -> k rows 0..7 (+4*i)
    const int bnc  = (tid & 31) * 4;

    float4 ra[8], rb[8];

    auto gload = [&](int k0) {
#pragma unroll
        for (int i = 0; i < 8; ++i)
            ra[i] = *(const float4*)(A + (size_t)(bm + arow + i * 16) * K + k0 + akc);
#pragma unroll
        for (int i = 0; i < 8; ++i)
            rb[i] = *(const float4*)(W + (size_t)(k0 + bkr + i * 4) * N + bn + bnc);
    };
    auto lstore = [&](int buf) {
#pragma unroll
        for (int i = 0; i < 8; ++i) {
            __bf16* d = &As[buf][(arow + i * 16) * 72 + akc];
            d[0] = (__bf16)ra[i].x; d[1] = (__bf16)ra[i].y;
            d[2] = (__bf16)ra[i].z; d[3] = (__bf16)ra[i].w;
        }
#pragma unroll
        for (int i = 0; i < 8; ++i) {
            int kr = bkr + i * 4;
            Bs[buf][(bnc + 0) * 72 + kr] = (__bf16)rb[i].x;
            Bs[buf][(bnc + 1) * 72 + kr] = (__bf16)rb[i].y;
            Bs[buf][(bnc + 2) * 72 + kr] = (__bf16)rb[i].z;
            Bs[buf][(bnc + 3) * 72 + kr] = (__bf16)rb[i].w;
        }
    };

    v8f acc[2][4];
#pragma unroll
    for (int i = 0; i < 2; ++i)
#pragma unroll
        for (int j = 0; j < 4; ++j) acc[i][j] = vzero8();

    gload(0);
    lstore(0);
    __syncthreads();

    for (int kt = 0; kt < nk; ++kt) {
        const int cur = kt & 1;
        // kick off next tile's global loads before computing (latency hiding)
        if (kt + 1 < nk) gload((kt + 1) << 6);
        // keep weights two tiles ahead warm in cache (global_prefetch_b8)
        if (kt + 2 < nk) {
            __builtin_prefetch(W + (size_t)((kt + 2) << 6) * N + bn + bnc, 0, 1);
            __builtin_prefetch(A + (size_t)(bm + arow) * K + ((kt + 2) << 6) + akc, 0, 1);
        }

#pragma unroll
        for (int kk = 0; kk < 2; ++kk) {
            const int akoff = kk * 32 + half * 8;
            v16bf afrag[2];
#pragma unroll
            for (int mt = 0; mt < 2; ++mt) {
                const __bf16* p = &As[cur][(wm * 32 + mt * 16 + l15) * 72 + akoff];
                afrag[mt] = load2x8(p, p + 16);
            }
            const int bkoff = kk * 32 + half * 8;
#pragma unroll
            for (int nt = 0; nt < 4; ++nt) {
                const __bf16* p = &Bs[cur][(wn * 64 + nt * 16 + l15) * 72 + bkoff];
                v16bf bfrag = load2x8(p, p + 16);
#pragma unroll
                for (int mt = 0; mt < 2; ++mt)
                    acc[mt][nt] = __builtin_amdgcn_wmma_f32_16x16x32_bf16(
                        false, afrag[mt], false, bfrag, (short)0, acc[mt][nt], false, false);
            }
        }
        if (kt + 1 < nk) lstore(cur ^ 1);
        __syncthreads();
    }

    // --- epilogue: C layout VGPR r -> row r (lanes 0-15) / r+8 (lanes 16-31)
#pragma unroll
    for (int mt = 0; mt < 2; ++mt)
#pragma unroll
        for (int nt = 0; nt < 4; ++nt) {
            int col = bn + wn * 64 + nt * 16 + l15;
            float bv = bias[col];
#pragma unroll
            for (int r = 0; r < 8; ++r) {
                int row = bm + wm * 32 + mt * 16 + r + half * 8;
                float v = acc[mt][nt][r] + bv;
                if (relu) v = fmaxf(v, 0.0f);
                C[(size_t)row * N + col] = v;
            }
        }
}

// ---------------------------------------------------------------------------
// Pack [B,S,D] fp32 -> [B,H,S,64] bf16, with scale (Q gets 1/sqrt(depth)).
// ---------------------------------------------------------------------------
__global__ __launch_bounds__(256)
void pack_heads_kernel(const float* __restrict__ src, __bf16* __restrict__ dst,
                       float scale)
{
    size_t i = (size_t)blockIdx.x * 256 + threadIdx.x;   // over B*S*D
    int d = (int)(i & 63);
    int h = (int)((i >> 6) & 15);
    int s = (int)((i >> 10) & 2047);
    int b = (int)(i >> 21);
    size_t o = (((size_t)(b * HEADS + h)) * SEQ + s) * DEPTH + d;
    dst[o] = (__bf16)(src[i] * scale);
}

// ---------------------------------------------------------------------------
// Flash attention.  grid = (S/128, B*H), block = 256 (8 waves, 16 q-rows each).
// Q,K,V bf16 [B,H,S,64]; mask fp32 [B,S]; output fp32 [B,S,D] (heads merged).
// Scale 1/8 folded into Q at pack time.  K tile staged with the CDNA5 async
// global->LDS path (ASYNCcnt); V staged via VGPRs (needs transpose).
// ---------------------------------------------------------------------------
__global__ __launch_bounds__(256)
void flash_attn_kernel(const __bf16* __restrict__ Q, const __bf16* __restrict__ Kg,
                       const __bf16* __restrict__ Vg, const float* __restrict__ mask,
                       float* __restrict__ O)
{
    __shared__ __align__(16) __bf16 Ks[32 * 72];       // [key][d], stride 72
    __shared__ __align__(16) __bf16 Vs[64 * 40];       // [d][key] (transposed), stride 40
    __shared__ __align__(16) __bf16 Ps[8 * 16 * 40];   // per-wave P tile [row][key]

    const int tid  = threadIdx.x;
    const int lane = tid & 31;
    const int wave = tid >> 5;
    const int half = lane >> 4;
    const int l15  = lane & 15;
    const int bh   = blockIdx.y;
    const int b    = bh >> 4;
    const int h    = bh & 15;
    const int qbase = blockIdx.x * 128 + wave * 16;
    const size_t head_base = (size_t)bh * SEQ * DEPTH;

    // per-thread staging coordinates: one 16B chunk of the 32x64 tile
    const int srow = tid >> 3;         // 0..31 (key)
    const int scc  = (tid & 7) * 8;    // 0..56 (d)
    // LDS byte offset of this thread's K destination (flat-LDS aperture: low 32
    // bits of the generic shared-pointer are the wave-relative LDS address)
    const unsigned kdst = (unsigned)(size_t)&Ks[srow * 72 + scc];

    // Q fragments for the whole tile (depth 64 = two 16x32 A-frags)
    v16bf aq[2];
    {
        const int qrow = qbase + l15;
        const int koff = half * 8;
#pragma unroll
        for (int dc = 0; dc < 2; ++dc) {
            const __bf16* p = Q + head_base + (size_t)qrow * DEPTH + dc * 32 + koff;
            aq[dc] = load2x8(p, p + 16);
        }
    }

    float m_i[8], l_i[8];
#pragma unroll
    for (int r = 0; r < 8; ++r) { m_i[r] = -1e30f; l_i[r] = 0.0f; }
    v8f ctx[4];
#pragma unroll
    for (int nv = 0; nv < 4; ++nv) ctx[nv] = vzero8();

    __bf16* myP = &Ps[wave * 16 * 40];
    const int bkoff = half * 8;

    for (int kt = 0; kt < SEQ; kt += 32) {
        __syncthreads();
        // --- async copy K tile [32 x 64] global -> LDS (16B per lane)
        {
            unsigned long long kga =
                (unsigned long long)(size_t)(Kg + head_base + (size_t)(kt + srow) * DEPTH + scc);
            asm volatile("global_load_async_to_lds_b128 %0, %1, off"
                         :: "v"(kdst), "v"(kga) : "memory");
        }
        // --- stage V tile transposed through VGPRs: Vs[d][key]
        {
            v8bf vv = *(const v8bf*)(Vg + head_base + (size_t)(kt + srow) * DEPTH + scc);
#pragma unroll
            for (int j = 0; j < 8; ++j) Vs[(scc + j) * 40 + srow] = vv[j];
        }
        asm volatile("s_wait_asynccnt 0" ::: "memory");
        __syncthreads();

        // logits 16x32 = Q(16x64) x K^T(64x32), two key-tiles of 16
        v8f s[2];
#pragma unroll
        for (int ntk = 0; ntk < 2; ++ntk) {
            const int key = ntk * 16 + l15;
            v16bf b0 = load2x8(&Ks[key * 72 +  0 + bkoff], &Ks[key * 72 +  0 + bkoff + 16]);
            v16bf b1 = load2x8(&Ks[key * 72 + 32 + bkoff], &Ks[key * 72 + 32 + bkoff + 16]);
            v8f c = vzero8();
            c = __builtin_amdgcn_wmma_f32_16x16x32_bf16(false, aq[0], false, b0, (short)0, c, false, false);
            c = __builtin_amdgcn_wmma_f32_16x16x32_bf16(false, aq[1], false, b1, (short)0, c, false, false);
            // additive mask (per key column, same for all rows)
            float mv = -1e9f * mask[(size_t)b * SEQ + kt + key];
#pragma unroll
            for (int r = 0; r < 8; ++r) c[r] += mv;
            s[ntk] = c;
        }

        // online softmax over the 32 keys; row r lives in lanes of its 16-group
        float alpha[8];
#pragma unroll
        for (int r = 0; r < 8; ++r) {
            float mx = fmaxf(s[0][r], s[1][r]);
#pragma unroll
            for (int off = 8; off >= 1; off >>= 1)
                mx = fmaxf(mx, __shfl_xor(mx, off, 32));
            float m_new = fmaxf(m_i[r], mx);
            alpha[r] = __expf(m_i[r] - m_new);
            m_i[r] = m_new;
            float p0 = __expf(s[0][r] - m_new);
            float p1 = __expf(s[1][r] - m_new);
            float rs = p0 + p1;
#pragma unroll
            for (int off = 8; off >= 1; off >>= 1)
                rs += __shfl_xor(rs, off, 32);
            l_i[r] = l_i[r] * alpha[r] + rs;
            // write P as bf16 to this wave's LDS tile (C-layout -> row-major)
            int rr = r + half * 8;
            myP[rr * 40 +      l15] = (__bf16)p0;
            myP[rr * 40 + 16 + l15] = (__bf16)p1;
        }
        // rescale running context
#pragma unroll
        for (int nv = 0; nv < 4; ++nv)
#pragma unroll
            for (int r = 0; r < 8; ++r) ctx[nv][r] *= alpha[r];

        // same-wave LDS store -> load handoff
        asm volatile("s_wait_dscnt 0" ::: "memory");

        // A-frag of P (16 rows x 32 keys)
        const __bf16* pp = &myP[l15 * 40 + half * 8];
        v16bf ap = load2x8(pp, pp + 16);

        // ctx(16x64) += P(16x32) x V(32x64): four 16-wide d tiles
#pragma unroll
        for (int nv = 0; nv < 4; ++nv) {
            const __bf16* vp = &Vs[(nv * 16 + l15) * 40 + bkoff];
            v16bf bv = load2x8(vp, vp + 16);
            ctx[nv] = __builtin_amdgcn_wmma_f32_16x16x32_bf16(
                false, ap, false, bv, (short)0, ctx[nv], false, false);
        }
    }

    // normalize and write merged-head output [B,S,D]
#pragma unroll
    for (int r = 0; r < 8; ++r) {
        float inv = 1.0f / l_i[r];
        int row = qbase + r + half * 8;
        size_t base = ((size_t)b * SEQ + row) * D_MODEL + h * DEPTH;
#pragma unroll
        for (int nv = 0; nv < 4; ++nv)
            O[base + nv * 16 + l15] = ctx[nv][r] * inv;
    }
}

// ---------------------------------------------------------------------------
// Fused residual-add + LayerNorm over rows of 1024.  One block per row.
// ---------------------------------------------------------------------------
__global__ __launch_bounds__(256)
void add_ln_kernel(const float* __restrict__ X, const float* __restrict__ Y,
                   const float* __restrict__ g, const float* __restrict__ bta,
                   float* __restrict__ out)
{
    __shared__ float2 red[8];
    const int row = blockIdx.x, tid = threadIdx.x;
    const int lane = tid & 31, wave = tid >> 5;
    const float* xr = X + (size_t)row * D_MODEL;
    const float* yr = Y + (size_t)row * D_MODEL;

    float v[4]; float s = 0.0f, s2 = 0.0f;
#pragma unroll
    for (int i = 0; i < 4; ++i) {
        int c = tid + i * 256;
        v[i] = xr[c] + yr[c];
        s += v[i]; s2 += v[i] * v[i];
    }
#pragma unroll
    for (int off = 16; off >= 1; off >>= 1) {
        s  += __shfl_xor(s,  off, 32);
        s2 += __shfl_xor(s2, off, 32);
    }
    if (lane == 0) red[wave] = make_float2(s, s2);
    __syncthreads();
    float ts = 0.0f, ts2 = 0.0f;
#pragma unroll
    for (int wI = 0; wI < 8; ++wI) { ts += red[wI].x; ts2 += red[wI].y; }
    float mu  = ts * (1.0f / D_MODEL);
    float var = ts2 * (1.0f / D_MODEL) - mu * mu;
    float inv = rsqrtf(var + 1e-6f);
#pragma unroll
    for (int i = 0; i < 4; ++i) {
        int c = tid + i * 256;
        out[(size_t)row * D_MODEL + c] = (v[i] - mu) * inv * g[c] + bta[c];
    }
}

// ---------------------------------------------------------------------------
// Orchestration
// ---------------------------------------------------------------------------
extern "C" void kernel_launch(void* const* d_in, const int* in_sizes, int n_in,
                              void* d_out, int out_size, void* d_ws, size_t ws_size,
                              hipStream_t stream) {
    const float* x     = (const float*)d_in[0];
    const float* mask  = (const float*)d_in[1];
    const float* wq_w  = (const float*)d_in[2];
    const float* wq_b  = (const float*)d_in[3];
    const float* wk_w  = (const float*)d_in[4];
    const float* wk_b  = (const float*)d_in[5];
    const float* wv_w  = (const float*)d_in[6];
    const float* wv_b  = (const float*)d_in[7];
    const float* wo_w  = (const float*)d_in[8];
    const float* wo_b  = (const float*)d_in[9];
    const float* fc1_w = (const float*)d_in[10];
    const float* fc1_b = (const float*)d_in[11];
    const float* fc2_w = (const float*)d_in[12];
    const float* fc2_b = (const float*)d_in[13];
    const float* ln1_g = (const float*)d_in[14];
    const float* ln1_b = (const float*)d_in[15];
    const float* ln2_g = (const float*)d_in[16];
    const float* ln2_b = (const float*)d_in[17];
    float* out = (float*)d_out;

    char* w = (char*)d_ws;
    float*  t0   = (float*)w;  w += (size_t)MROWS * D_MODEL * 4;
    float*  t1   = (float*)w;  w += (size_t)MROWS * D_MODEL * 4;
    float*  t2   = (float*)w;  w += (size_t)MROWS * D_MODEL * 4;
    float*  tffn = (float*)w;  w += (size_t)MROWS * DFF * 4;
    __bf16* Qh   = (__bf16*)w; w += (size_t)MROWS * D_MODEL * 2;
    __bf16* Kh   = (__bf16*)w; w += (size_t)MROWS * D_MODEL * 2;
    __bf16* Vh   = (__bf16*)w; w += (size_t)MROWS * D_MODEL * 2;

    dim3 blk(256);
    dim3 gD(MROWS / 128, D_MODEL / 128);   // 32 x 8
    dim3 gF(MROWS / 128, DFF / 128);       // 32 x 32
    const int nPack = (MROWS * D_MODEL) / 256;

    // Q/K/V projections + head packing (scale 1/sqrt(64) folded into Q)
    gemm_bf16_kernel<<<gD, blk, 0, stream>>>(x, wq_w, wq_b, t0, MROWS, D_MODEL, D_MODEL, 0);
    pack_heads_kernel<<<nPack, blk, 0, stream>>>(t0, Qh, 0.125f);
    gemm_bf16_kernel<<<gD, blk, 0, stream>>>(x, wk_w, wk_b, t0, MROWS, D_MODEL, D_MODEL, 0);
    pack_heads_kernel<<<nPack, blk, 0, stream>>>(t0, Kh, 1.0f);
    gemm_bf16_kernel<<<gD, blk, 0, stream>>>(x, wv_w, wv_b, t0, MROWS, D_MODEL, D_MODEL, 0);
    pack_heads_kernel<<<nPack, blk, 0, stream>>>(t0, Vh, 1.0f);

    // flash attention -> merged-head ctx in t0
    flash_attn_kernel<<<dim3(SEQ / 128, BATCH * HEADS), blk, 0, stream>>>(Qh, Kh, Vh, mask, t0);

    // output projection, residual + LN1
    gemm_bf16_kernel<<<gD, blk, 0, stream>>>(t0, wo_w, wo_b, t1, MROWS, D_MODEL, D_MODEL, 0);
    add_ln_kernel<<<MROWS, blk, 0, stream>>>(x, t1, ln1_g, ln1_b, t2);   // out1 in t2

    // FFN
    gemm_bf16_kernel<<<gF, blk, 0, stream>>>(t2, fc1_w, fc1_b, tffn, MROWS, DFF, D_MODEL, 1);
    gemm_bf16_kernel<<<gD, blk, 0, stream>>>(tffn, fc2_w, fc2_b, t0, MROWS, D_MODEL, DFF, 0);

    // residual + LN2 -> output
    add_ln_kernel<<<MROWS, blk, 0, stream>>>(t2, t0, ln2_g, ln2_b, out);
}
